// Attention_68393059221880
// MI455X (gfx1250) — matrix operand order repaired and checked by reading, not compiled, not run on hardware
//
#include <hip/hip_runtime.h>

// ---------------------------------------------------------------------------
// Flash-attention + fused FC for MI455X (gfx1250, wave32, WMMA bf16, TDM).
//   attn = softmax(Q K^T / 8); O = attn V; Out = O W^T + b
// Shapes: N=4, P=4096, d_k=64, d_v=256.
// One workgroup = 16 query rows of one batch; 4 waves, each wave owns a
// 64-wide d_v slice. Online softmax (flash) over 128 chunks of 32 keys.
// V chunks are staged into LDS by the Tensor Data Mover (one quarter per
// wave), with TDM padding giving a 257-dword row pitch (bank-conflict-free
// column reads). DMA overlaps the QK^T/softmax phase.
// ---------------------------------------------------------------------------

#define NBATCH 4
#define SEQ    4096
#define DK     64
#define DV     256
#define ROWS   16          // Q rows per workgroup
#define CK     32          // keys per chunk
#define VFP    257         // f32 row pitch (DV + 1 pad dword via TDM padding)
#define PLP    40          // LDS pitch (keys) for per-wave P tile
#define OLP    264         // LDS pitch (channels) for O tile (pad 8)

typedef __attribute__((ext_vector_type(16))) __bf16       bf16x16;
typedef __attribute__((ext_vector_type(8)))  float        f32x8;
typedef __attribute__((ext_vector_type(4)))  unsigned int u32x4;
typedef __attribute__((ext_vector_type(8)))  int          i32x8;
typedef __attribute__((ext_vector_type(4)))  int          i32x4;

__device__ __forceinline__ float half_rmax(float x) {
  x = fmaxf(x, __shfl_xor(x, 1, 32));
  x = fmaxf(x, __shfl_xor(x, 2, 32));
  x = fmaxf(x, __shfl_xor(x, 4, 32));
  x = fmaxf(x, __shfl_xor(x, 8, 32));
  return x;
}
__device__ __forceinline__ float half_rsum(float x) {
  x += __shfl_xor(x, 1, 32);
  x += __shfl_xor(x, 2, 32);
  x += __shfl_xor(x, 4, 32);
  x += __shfl_xor(x, 8, 32);
  return x;
}

// TDM: DMA 8 key rows (8 x 256 f32 = 2048 dwords, contiguous in global) into
// LDS at lds_off, inserting 1 pad dword after every 256 dwords (row pitch 257).
__device__ __forceinline__ void tdm_load_quarter(const float* gsrc,
                                                 unsigned lds_off) {
  const unsigned long long ga = (unsigned long long)(size_t)gsrc;
  u32x4 g0;
  g0[0] = 1u;                                   // count=1, user descriptor
  g0[1] = lds_off;                              // lds_addr (bytes)
  g0[2] = (unsigned)(ga & 0xFFFFFFFFull);       // global_addr[31:0]
  g0[3] = (unsigned)((ga >> 32) & 0x1FFFFFFu)   // global_addr[56:32]
        | (2u << 30);                           // type=2 ("image")
  i32x8 g1;
  g1[0] = (2 << 16)                             // data_size = 4B
        | (1 << 20)                             // pad_enable
        | (7 << 22);                            // pad_interval = 256 dwords
                                                // pad_amount = 0 -> 1 dword
  g1[1] = 0;                                    // tensor_dim0 lo16 in [31:16]
  g1[2] = 16 | (1 << 16);                       // tensor_dim0=1<<20; dim1=1
  g1[3] = (int)(2048u << 16);                   // tile_dim0 = 2048 dwords
  g1[4] = 0;                                    // tile_dim1/2 unused (1-D)
  g1[5] = 1 << 20;                              // tensor_dim0_stride
  g1[6] = 0;
  g1[7] = 0;
  i32x4 gz4 = {0, 0, 0, 0};
#if __clang_major__ >= 23
  i32x8 gz8 = {0, 0, 0, 0, 0, 0, 0, 0};
  __builtin_amdgcn_tensor_load_to_lds(g0, g1, gz4, gz4, gz8, 0);
#else
  __builtin_amdgcn_tensor_load_to_lds(g0, g1, gz4, gz4, 0);
#endif
}

__global__ __launch_bounds__(128)
void flash_attn_fc_kernel(const float* __restrict__ Ksrc,
                          const float* __restrict__ Vsrc,
                          const float* __restrict__ Q,
                          const float* __restrict__ Wfc,
                          const float* __restrict__ bfc,
                          float* __restrict__ Out)
{
  __shared__ float  Vf[CK * VFP];          // V chunk f32: [key][c], pitch 257
  __shared__ __bf16 Plds[4][ROWS * PLP];   // per-wave P tile: [row][key]
  __shared__ __bf16 Olds[ROWS * OLP];      // O tile: [row][c]

  const int tid  = threadIdx.x;
  const int lane = tid & 31;
  const int wv   = tid >> 5;               // wave 0..3
  const int l16  = lane & 15;
  const int hi   = lane >> 4;              // lane half: 0 or 1

  const int blk  = blockIdx.x;
  const int n    = blk >> 8;               // SEQ/ROWS = 256 row tiles / batch
  const int row0 = (blk & 255) << 4;

  const unsigned vf_base =
      (unsigned)(size_t)(&Vf[0]) + (unsigned)(wv * 8 * VFP * 4);

  // ---- Q tile -> two A-fragments (16x32 bf16, contiguous k = 16*hi + e) ----
  bf16x16 qa0, qa1;
  {
    const int k0 = hi * 16;
    const float* qrow = Q + ((size_t)(n * SEQ + row0 + l16)) * DK;
#pragma unroll
    for (int e = 0; e < 16; ++e) qa0[e] = (__bf16)qrow[k0 + e];
#pragma unroll
    for (int e = 0; e < 16; ++e) qa1[e] = (__bf16)qrow[32 + k0 + e];
  }

  const f32x8 zero = {};
  f32x8 acc[4];
#pragma unroll
  for (int t = 0; t < 4; ++t) acc[t] = zero;

  float m_i[8], l_i[8];
#pragma unroll
  for (int r = 0; r < 8; ++r) { m_i[r] = -1e30f; l_i[r] = 0.0f; }

  __bf16* Pw = Plds[wv];

  for (int j = 0; j < SEQ / CK; ++j) {
    const int kb = j * CK;

    // Previous chunk's readers must be done before the TDM overwrites Vf.
    __syncthreads();

    // ---- issue TDM: this wave's 8-key quarter of the V chunk ----
    tdm_load_quarter(Vsrc + ((size_t)(n * SEQ + kb + wv * 8)) * DV, vf_base);

    // gfx1250 prefetch hint for next K chunk (global_prefetch_b8)
    if (j + 1 < SEQ / CK) {
      const float* nk = Ksrc + ((size_t)(n * SEQ + kb + CK + lane)) * DK;
      __builtin_prefetch(nk, 0, 1);
    }

    // ---- S = (Q K^T) for 32 keys: two 16x16 f32 tiles (overlaps DMA) ----
    f32x8 s0, s1;
    {
      const float* krA = Ksrc + ((size_t)(n * SEQ + kb + l16)) * DK + hi * 16;
      const float* krB = Ksrc + ((size_t)(n * SEQ + kb + 16 + l16)) * DK + hi * 16;
      bf16x16 b00, b01, b10, b11;
#pragma unroll
      for (int e = 0; e < 16; ++e) b00[e] = (__bf16)krA[e];
#pragma unroll
      for (int e = 0; e < 16; ++e) b01[e] = (__bf16)krA[32 + e];
#pragma unroll
      for (int e = 0; e < 16; ++e) b10[e] = (__bf16)krB[e];
#pragma unroll
      for (int e = 0; e < 16; ++e) b11[e] = (__bf16)krB[32 + e];
      s0 = __builtin_amdgcn_wmma_f32_16x16x32_bf16(false, qa0, false, b00,
                                                   (short)0, zero, false, false);
      s0 = __builtin_amdgcn_wmma_f32_16x16x32_bf16(false, qa1, false, b01,
                                                   (short)0, s0, false, false);
      s1 = __builtin_amdgcn_wmma_f32_16x16x32_bf16(false, qa0, false, b10,
                                                   (short)0, zero, false, false);
      s1 = __builtin_amdgcn_wmma_f32_16x16x32_bf16(false, qa1, false, b11,
                                                   (short)0, s1, false, false);
    }

    // ---- online softmax (per row; rows live in 16-lane halves) ----
#pragma unroll
    for (int r = 0; r < 8; ++r) {
      const float a0 = s0[r] * 0.125f;       // / sqrt(64)
      const float a1 = s1[r] * 0.125f;
      const float cm   = half_rmax(fmaxf(a0, a1));
      const float mnew = fmaxf(m_i[r], cm);
      const float alpha = __expf(m_i[r] - mnew);
      const float p0 = __expf(a0 - mnew);
      const float p1 = __expf(a1 - mnew);
      l_i[r] = l_i[r] * alpha + half_rsum(p0 + p1);
      m_i[r] = mnew;
#pragma unroll
      for (int t = 0; t < 4; ++t) acc[t][r] *= alpha;
      const int row = r + hi * 8;            // C-layout row
      Pw[row * PLP + l16]      = (__bf16)p0;
      Pw[row * PLP + 16 + l16] = (__bf16)p1;
    }

    // ---- P (C-layout f32) -> A-layout bf16 via wave-private LDS ----
    bf16x16 pa;
    {
      const __bf16* prow = Pw + l16 * PLP;
      const int k0 = hi * 16;
#pragma unroll
      for (int e = 0; e < 16; ++e) pa[e] = prow[k0 + e];
    }

    // ---- wait for the DMA'd V chunk, publish across waves ----
    __builtin_amdgcn_s_wait_tensorcnt(0);
    __syncthreads();

    // ---- O += P * V_chunk (wave's 64-wide d_v slice) ----
#pragma unroll
    for (int t = 0; t < 4; ++t) {
      const int c = wv * 64 + t * 16 + l16;
      const float* vc = Vf + (hi * 16) * VFP + c;   // key = e + 16*hi
      bf16x16 bv;
#pragma unroll
      for (int e = 0; e < 16; ++e) bv[e] = (__bf16)vc[e * VFP];
      acc[t] = __builtin_amdgcn_wmma_f32_16x16x32_bf16(false, pa, false, bv,
                                                       (short)0, acc[t],
                                                       false, false);
    }
  }

  // ---- normalize and stage O tile (bf16) for the fused FC ----
#pragma unroll
  for (int r = 0; r < 8; ++r) {
    const float inv = 1.0f / l_i[r];
    const int row = r + hi * 8;
#pragma unroll
    for (int t = 0; t < 4; ++t)
      Olds[row * OLP + wv * 64 + t * 16 + l16] = (__bf16)(acc[t][r] * inv);
  }
  __syncthreads();

  // ---- FC: Out[16, o-slice] = O[16,256] @ W[o-slice,:]^T + b ----
  f32x8 facc[4];
#pragma unroll
  for (int t = 0; t < 4; ++t) facc[t] = zero;

#pragma unroll
  for (int kk = 0; kk < DV / 32; ++kk) {     // 8 K-steps of 32 channels
    bf16x16 oa;
    {
      const __bf16* orow = Olds + l16 * OLP + kk * 32;
      const int k0 = hi * 16;
#pragma unroll
      for (int e = 0; e < 16; ++e) oa[e] = orow[k0 + e];
    }
#pragma unroll
    for (int t = 0; t < 4; ++t) {
      const int o = wv * 64 + t * 16 + l16;
      const float* wr = Wfc + (size_t)o * DV + kk * 32 + hi * 16;
      bf16x16 bw;
#pragma unroll
      for (int e = 0; e < 16; ++e) bw[e] = (__bf16)wr[e];
      facc[t] = __builtin_amdgcn_wmma_f32_16x16x32_bf16(false, oa, false, bw,
                                                        (short)0, facc[t],
                                                        false, false);
    }
  }

  // ---- bias + store f32 ----
#pragma unroll
  for (int t = 0; t < 4; ++t) {
    const int o = wv * 64 + t * 16 + l16;
    const float bias = bfc[o];
    float* orow = Out + ((size_t)(n * SEQ + row0)) * DV + o;
#pragma unroll
    for (int r = 0; r < 8; ++r) {
      const int row = r + hi * 8;
      orow[(size_t)row * DV] = facc[t][r] + bias;
    }
  }
}

extern "C" void kernel_launch(void* const* d_in, const int* in_sizes, int n_in,
                              void* d_out, int out_size, void* d_ws,
                              size_t ws_size, hipStream_t stream) {
  (void)in_sizes; (void)n_in; (void)out_size; (void)d_ws; (void)ws_size;
  const float* Ksrc = (const float*)d_in[0];   // k_src [4,4096,64]
  const float* Vsrc = (const float*)d_in[1];   // v_src [4,4096,256]
  const float* Q    = (const float*)d_in[2];   // q_tgr [4,4096,64]
  const float* Wfc  = (const float*)d_in[3];   // W_fc  [256,256]
  const float* bfc  = (const float*)d_in[4];   // b_fc  [256]
  float* Out = (float*)d_out;                  // [4,4096,256] f32

  dim3 grid(NBATCH * (SEQ / ROWS));            // 1024 workgroups
  dim3 block(128);                             // 4 waves (wave32)
  flash_attn_fc_kernel<<<grid, block, 0, stream>>>(Ksrc, Vsrc, Q, Wfc, bfc, Out);
}
